// MultiHeadAttention_12687333393018
// MI455X (gfx1250) — matrix operand order repaired and checked by reading, not compiled, not run on hardware
//
#include <hip/hip_runtime.h>

typedef __attribute__((ext_vector_type(16))) _Float16 v16h;
typedef __attribute__((ext_vector_type(8)))  _Float16 v8h;
typedef __attribute__((ext_vector_type(2)))  __fp16   v2hf;  // cvt_pkrtz result type
typedef __attribute__((ext_vector_type(4)))  __fp16   v4hf;
typedef __attribute__((ext_vector_type(8)))  float    v8f;

#define S_LEN  2048
#define DIMS   1024
#define HEADS  16
#define DEPTH  64
#define BQ     64     // q rows per workgroup (4 waves x 16)
#define BK     64     // keys per iteration (4 x 16-key subtiles)
#define NWAVES 4

// Build a v16h A/B operand from two 16-byte LDS chunks.
__device__ __forceinline__ v16h load_pair(const _Float16* p0, const _Float16* p1) {
  v8h a = *(const v8h*)p0;
  v8h b = *(const v8h*)p1;
  v16h r;
#pragma unroll
  for (int i = 0; i < 8; ++i) { r[i] = a[i]; r[i + 8] = b[i]; }
  return r;
}

// float4 -> 4 packed halves via v_cvt_pk_rtz_f16_f32
__device__ __forceinline__ v4hf cvt4(float4 f) {
  v2hf p0 = __builtin_amdgcn_cvt_pkrtz(f.x, f.y);
  v2hf p1 = __builtin_amdgcn_cvt_pkrtz(f.z, f.w);
  v4hf h;
  h[0] = p0[0]; h[1] = p0[1]; h[2] = p1[0]; h[3] = p1[1];
  return h;
}

// packed half2 max (VOP3P V_PK_MAX_NUM_F16)
__device__ __forceinline__ unsigned pkmax(unsigned a, unsigned b) {
  unsigned d;
  asm("v_pk_max_num_f16 %0, %1, %2" : "=v"(d) : "v"(a), "v"(b));
  return d;
}

__global__ __launch_bounds__(128) void flash_attn_fwd(
    const float* __restrict__ q, const float* __restrict__ k,
    const float* __restrict__ v, float* __restrict__ out) {
  __shared__ _Float16 sQ[BQ][DEPTH];        // 8 KB
  __shared__ _Float16 sK[BK][DEPTH];        // 8 KB
  __shared__ _Float16 sVt[DEPTH][BK];       // 8 KB (V transposed)
  __shared__ _Float16 sP[NWAVES][16][BK];   // 8 KB (P staging, per wave)

  const int tid  = threadIdx.x;
  const int lane = tid & 31;
  const int wave = tid >> 5;
  const int nQT  = S_LEN / BQ;                       // 32
  const int qt    = blockIdx.x % nQT;
  const int head  = (blockIdx.x / nQT) % HEADS;
  const int batch = blockIdx.x / (nQT * HEADS);

  const int q0 = qt * BQ;
  const size_t base = (size_t)batch * S_LEN * DIMS + (size_t)head * DEPTH;
  const float LOG2E = 1.4426950408889634f;

  // ---- load Q tile to LDS, pre-scaled by log2(e) so S is in exp2 domain ----
#pragma unroll
  for (int i = 0; i < (BQ * DEPTH / 4) / 128; ++i) {   // 8
    int idx = tid + i * 128;
    int r = idx >> 4, c4 = idx & 15;
    float4 f = *(const float4*)&q[base + (size_t)(q0 + r) * DIMS + c4 * 4];
    f.x *= LOG2E; f.y *= LOG2E; f.z *= LOG2E; f.w *= LOG2E;
    *(v4hf*)&sQ[r][c4 * 4] = cvt4(f);
  }
  __syncthreads();

  // ---- per-wave Q A-matrix registers (16x64 = two 16x32 A tiles) ----
  const int rowA = lane & 15;
  const int hb = (lane < 16) ? 0 : 8;   // A-layout K base (lo: K0..7/16..23, hi: K8..15/24..31)
  const int hk = (lane < 16) ? 0 : 16;  // contiguous K base, unused (B loads use hb, same K order as A)
  const int hr = (lane < 16) ? 0 : 8;   // C-layout row base
  const _Float16* qrow = &sQ[wave * 16 + rowA][0];
  const v16h qa0 = load_pair(qrow + 0  + hb, qrow + 16 + hb);
  const v16h qa1 = load_pair(qrow + 32 + hb, qrow + 48 + hb);

  // all-ones B matrix: P x ones -> row sums of P, replicated into every C column
  v16h vone;
#pragma unroll
  for (int i = 0; i < 16; ++i) vone[i] = (_Float16)1.0f;

  v8f o[4] = {{}, {}, {}, {}};
  v8f m8, l8;
#pragma unroll
  for (int j = 0; j < 8; ++j) { m8[j] = -1.0e30f; l8[j] = 0.0f; }

  for (int kb = 0; kb < S_LEN; kb += BK) {
    __syncthreads();
    // ---- cooperative stream of K (row-major, float4 + pk-cvt) ----
#pragma unroll
    for (int i = 0; i < (BK * DEPTH / 4) / 128; ++i) {  // 8
      int idx = tid + i * 128;
      int kr = idx >> 4, c4 = idx & 15;
      float4 f = *(const float4*)&k[base + (size_t)(kb + kr) * DIMS + c4 * 4];
      *(v4hf*)&sK[kr][c4 * 4] = cvt4(f);
    }
    // ---- V transposed: pack across key pairs (same d) -> single b32 store ----
#pragma unroll
    for (int i = 0; i < (BK / 2 * DEPTH) / 128; ++i) {  // 16
      int idx = tid + i * 128;
      int d = idx & 63, key = (idx >> 6) * 2;
      size_t g = base + (size_t)(kb + key) * DIMS + d;
      *(v2hf*)&sVt[d][key] = __builtin_amdgcn_cvt_pkrtz(v[g], v[g + DIMS]);
    }
    // ---- prefetch next K/V tile into cache while we compute ----
    if (kb + BK < S_LEN) {
#pragma unroll
      for (int i = 0; i < 2; ++i) {
        int idx = tid + i * 128;
        int r = idx >> 2, c = (idx & 3) * 16;           // 64 rows x 4 chunks of 64B
        size_t g = base + (size_t)(kb + BK + r) * DIMS + c;
        __builtin_prefetch(&k[g], 0, 3);
        __builtin_prefetch(&v[g], 0, 3);
      }
    }
    __syncthreads();

    // ---- S = Q * K^T for four 16-key subtiles (fp32 accum, exp2 domain) ----
    v8f s[4] = {{}, {}, {}, {}};
#pragma unroll
    for (int t = 0; t < 4; ++t) {
      const _Float16* krp = &sK[t * 16 + (lane & 15)][0];
      v16h b;
      b    = load_pair(krp + hb,      krp + 16 + hb);       // d 0..31
      s[t] = __builtin_amdgcn_wmma_f32_16x16x32_f16(false, qa0, false, b, (short)0, s[t], false, false);
      b    = load_pair(krp + 32 + hb, krp + 48 + hb);       // d 32..63
      s[t] = __builtin_amdgcn_wmma_f32_16x16x32_f16(false, qa1, false, b, (short)0, s[t], false, false);
    }

    // ---- row max: fold 4 tiles, then packed-f16 butterfly over 16 lanes ----
    v8f t8;
#pragma unroll
    for (int j = 0; j < 8; ++j)
      t8[j] = fmaxf(fmaxf(s[0][j], s[1][j]), fmaxf(s[2][j], s[3][j]));

    unsigned pk4[4];
#pragma unroll
    for (int i = 0; i < 4; ++i)
      pk4[i] = __builtin_bit_cast(unsigned, __builtin_amdgcn_cvt_pkrtz(t8[2 * i], t8[2 * i + 1]));
#pragma unroll
    for (int off = 1; off < 16; off <<= 1) {
#pragma unroll
      for (int i = 0; i < 4; ++i)
        pk4[i] = pkmax(pk4[i], (unsigned)__shfl_xor((int)pk4[i], off, 32));
    }
    float mr[8];
#pragma unroll
    for (int i = 0; i < 4; ++i) {
      v2hf h = __builtin_bit_cast(v2hf, pk4[i]);
      mr[2 * i] = (float)h[0]; mr[2 * i + 1] = (float)h[1];
    }

    v8f mnew, alpha;
#pragma unroll
    for (int j = 0; j < 8; ++j) {
      mnew[j]  = fmaxf(m8[j], mr[j]);
      alpha[j] = exp2f(m8[j] - mnew[j]);
      m8[j]    = mnew[j];
    }
#pragma unroll
    for (int t = 0; t < 4; ++t)
#pragma unroll
      for (int j = 0; j < 8; ++j) s[t][j] = exp2f(s[t][j] - mnew[j]);

    // ---- stage P (C layout) -> LDS -> reload in A layout ----
#pragma unroll
    for (int t = 0; t < 4; ++t)
#pragma unroll
      for (int j = 0; j < 8; ++j)
        sP[wave][hr + j][t * 16 + (lane & 15)] = (_Float16)s[t][j];
    asm volatile("s_wait_dscnt 0" ::: "memory");  // intra-wave cross-lane LDS RAW
    const _Float16* prow = &sP[wave][rowA][0];
    const v16h pa0 = load_pair(prow + hb,      prow + 16 + hb);   // keys 0..31
    const v16h pa1 = load_pair(prow + 32 + hb, prow + 48 + hb);   // keys 32..63

    // ---- row sums of P via WMMA against all-ones B ----
    v8f rs = {};
    rs = __builtin_amdgcn_wmma_f32_16x16x32_f16(false, pa0, false, vone, (short)0, rs, false, false);
    rs = __builtin_amdgcn_wmma_f32_16x16x32_f16(false, pa1, false, vone, (short)0, rs, false, false);

#pragma unroll
    for (int j = 0; j < 8; ++j) {
      l8[j] = l8[j] * alpha[j] + rs[j];
      o[0][j] *= alpha[j]; o[1][j] *= alpha[j]; o[2][j] *= alpha[j]; o[3][j] *= alpha[j];
    }

    // ---- O += P * V (4 output column tiles of 16, 2 key-chunks each) ----
#pragma unroll
    for (int t = 0; t < 4; ++t) {
      const _Float16* vrp = &sVt[t * 16 + (lane & 15)][0];
      v16h b;
      b    = load_pair(vrp + hb,      vrp + 16 + hb);        // keys 0..31
      o[t] = __builtin_amdgcn_wmma_f32_16x16x32_f16(false, pa0, false, b, (short)0, o[t], false, false);
      b    = load_pair(vrp + 32 + hb, vrp + 48 + hb);        // keys 32..63
      o[t] = __builtin_amdgcn_wmma_f32_16x16x32_f16(false, pa1, false, b, (short)0, o[t], false, false);
    }
  }

  // ---- normalize and write out (same [B,S,DIMS] layout, head-column slab) ----
#pragma unroll
  for (int j = 0; j < 8; ++j) {
    const int row = q0 + wave * 16 + hr + j;
    const float inv = 1.0f / l8[j];
    const size_t rb = base + (size_t)row * DIMS + (lane & 15);
#pragma unroll
    for (int t = 0; t < 4; ++t) out[rb + t * 16] = o[t][j] * inv;
  }
}

extern "C" void kernel_launch(void* const* d_in, const int* in_sizes, int n_in,
                              void* d_out, int out_size, void* d_ws, size_t ws_size,
                              hipStream_t stream) {
  const float* q = (const float*)d_in[0];
  const float* k = (const float*)d_in[1];
  const float* v = (const float*)d_in[2];
  float* out = (float*)d_out;
  const int nbatch = in_sizes[0] / (S_LEN * DIMS);   // 2
  dim3 grid(nbatch * HEADS * (S_LEN / BQ));
  flash_attn_fwd<<<grid, 128, 0, stream>>>(q, k, v, out);
  (void)n_in; (void)out_size; (void)d_ws; (void)ws_size;
}
